// SblHmmLm_30459908063248
// MI455X (gfx1250) — matrix operand order repaired and checked by reading, not compiled, not run on hardware
//
#include <hip/hip_runtime.h>

typedef __attribute__((ext_vector_type(16))) __bf16 v16bf;
typedef __attribute__((ext_vector_type(8)))  __bf16 v8bf;
typedef __attribute__((ext_vector_type(8)))  float  v8f;

// Async global->LDS copy via inline asm (VGLOBAL encoding, ASYNCcnt-tracked).
#define USE_ASYNC_LDS 1
__device__ __forceinline__ void async_copy16(const float* g, float* l) {
    asm volatile("global_load_async_to_lds_b128 %0, %1, off"
                 :: "v"((__attribute__((address_space(3))) float*)l),
                    "v"((const __attribute__((address_space(1))) float*)g)
                 : "memory");
}
#if defined(__has_builtin)
#if __has_builtin(__builtin_amdgcn_s_wait_asynccnt)
#define WAIT_ASYNC() __builtin_amdgcn_s_wait_asynccnt(0)
#endif
#endif
#ifndef WAIT_ASYNC
#define WAIT_ASYNC() asm volatile("s_wait_asynccnt 0x0" ::: "memory")
#endif

namespace {
constexpr int BBATCH = 16;    // B
constexpr int TLEN   = 256;   // T
constexpr int VWORDS = 10000; // V
constexpr int CSTATE = 4096;  // C
constexpr int HDIM   = 256;   // H
constexpr int DDIM   = 256;   // D (= N = K of all big GEMMs)
constexpr int NCLUS  = 128;   // NC
constexpr int PITCH  = 40;    // LDS row pitch in bf16 elems (16B-aligned groups, conflict-free)
}

// ---------------------------------------------------------------------------
// Tiled WMMA GEMM, K = N = 256 fixed.  out = epilogue(A[M,256] @ Bw[256,256]).
//   mode 0: out = relu(acc + bias)
//   mode 1: out = res + relu(acc + bias)
//   mode 2: out = exp(acc - 0.5*sumsq(A-row))        (phi + exp fused)
// Block: 128 threads (4 waves), 32x32 output tile; K loop fully unrolled -> 8 WMMA.
// ---------------------------------------------------------------------------
__global__ __launch_bounds__(128) void wmma_gemm256(
    const float* __restrict__ A, const float* __restrict__ Bw,
    const float* __restrict__ bias, const float* __restrict__ res,
    float* __restrict__ out, int M, int mode)
{
    __shared__ __bf16 lA[32 * PITCH];   // [row][k]
    __shared__ __bf16 lBt[32 * PITCH];  // transposed: [n][k]
    __shared__ float  ssred[128];
    __shared__ float  ssrow[32];

    const int tid  = threadIdx.x;
    const int m0   = blockIdx.x * 32;
    const int n0   = blockIdx.y * 32;
    const int wave = tid >> 5;
    const int lane = tid & 31;
    const int wm   = wave >> 1;   // wave tile row (0..1)
    const int wn   = wave & 1;    // wave tile col (0..1)

    v8f acc = {};
    float ss = 0.f;

    const int ar  = tid >> 2;        // A row this thread stages (0..31)
    const int ac0 = (tid & 3) * 8;   // A col base (0,8,16,24)
    const int bn  = tid & 31;        // B col this thread stages (0..31)
    const int bk0 = (tid >> 5) * 8;  // B k base (0,8,16,24)
    // clamp row so the loads are always in-bounds & unconditional (vectorizable),
    // zero the contribution for padded rows via multiplicative mask
    const int   arow  = m0 + ar;
    const int   grow  = (arow < M) ? arow : (M - 1);
    const float zmask = (arow < M) ? 1.f : 0.f;

    // fragment source pointers (constant across K steps)
    const int fr = lane & 15;
    const int kh = lane >> 4;
    const __bf16* pa0 = &lA [(wm * 16 + fr) * PITCH + kh * 8];   // A: K = kh*8..+7
    const __bf16* pa1 = pa0 + 16;                                // A: K = 16+kh*8..+7
    const __bf16* pb0 = &lBt[(wn * 16 + fr) * PITCH + kh * 8];   // B: K = kh*8..+7
    const __bf16* pb1 = pb0 + 16;                                // B: K = 16+kh*8..+7

    #pragma unroll
    for (int kk = 0; kk < 256; kk += 32) {
        // ---- stage A tile (f32 -> bf16), accumulate row sum-of-squares in f32
        {
            const float* p = A + (size_t)grow * 256 + kk + ac0;
            v8bf apk;
            #pragma unroll
            for (int e = 0; e < 8; ++e) {
                float v = p[e] * zmask;          // unconditional contiguous loads
                ss += v * v;
                apk[e] = (__bf16)v;
            }
            *(v8bf*)&lA[ar * PITCH + ac0] = apk;   // one 16B ds_store
        }
        // ---- stage B tile transposed: lBt[n][k] = Bw[kk+k][n0+n]
        {
            v8bf bpk;
            #pragma unroll
            for (int e = 0; e < 8; ++e)            // coalesced 128B rows across lanes
                bpk[e] = (__bf16)Bw[(size_t)(kk + bk0 + e) * 256 + n0 + bn];
            *(v8bf*)&lBt[bn * PITCH + bk0] = bpk;  // one 16B ds_store
        }
        __syncthreads();

        v8bf alo = *(const v8bf*)pa0, ahi = *(const v8bf*)pa1;
        v8bf blo = *(const v8bf*)pb0, bhi = *(const v8bf*)pb1;
        v16bf af = __builtin_shufflevector(alo, ahi, 0,1,2,3,4,5,6,7,8,9,10,11,12,13,14,15);
        v16bf bf = __builtin_shufflevector(blo, bhi, 0,1,2,3,4,5,6,7,8,9,10,11,12,13,14,15);
        acc = __builtin_amdgcn_wmma_f32_16x16x32_bf16(false, af, false, bf,
                                                      (short)0, acc, false, false);
        __syncthreads();
    }

    // ---- reduce per-row sumsq (4 loader-threads per row)
    ssred[tid] = ss;
    __syncthreads();
    if (tid < 32)
        ssrow[tid] = ssred[tid*4] + ssred[tid*4+1] + ssred[tid*4+2] + ssred[tid*4+3];
    __syncthreads();

    // ---- epilogue + store (C/D layout: lane gives col, elem r gives row)
    const int col = n0 + wn * 16 + (lane & 15);
    #pragma unroll
    for (int r = 0; r < 8; ++r) {
        int rowL = wm * 16 + (lane >> 4) * 8 + r;
        int row  = m0 + rowL;
        if (row < M) {
            float v = acc[r];
            if (mode == 0)      v = fmaxf(v + bias[col], 0.f);
            else if (mode == 1) v = res[(size_t)row * 256 + col] + fmaxf(v + bias[col], 0.f);
            else                v = expf(v - 0.5f * ssrow[rowL]);
            out[(size_t)row * 256 + col] = v;
        }
    }
}

// ---------------------------------------------------------------------------
// fx chain (tiny): fx = res(res(start_emb@w0+b0, w1,b1,w2,b2), w3,b3,w4,b4)
// then EXPFX[d] = exp(fx . proj[:,d] - 0.5*||fx||^2).  One 256-thread block.
// ---------------------------------------------------------------------------
__global__ __launch_bounds__(256) void k_fx(
    const float* __restrict__ start_emb, const float* __restrict__ sw,
    const float* __restrict__ sb, const float* __restrict__ proj,
    float* __restrict__ expfx)
{
    __shared__ float A[256], Bv[256], R[256], red[256];
    const int t = threadIdx.x;
    float a;
    A[t] = start_emb[t];                       __syncthreads();
    a = 0.f; for (int k = 0; k < 256; ++k) a += A[k]  * sw[k*256 + t];
    Bv[t] = a + sb[t];                         __syncthreads();   // fx0
    R[t] = Bv[t];                              __syncthreads();
    a = 0.f; for (int k = 0; k < 256; ++k) a += Bv[k] * sw[65536   + k*256 + t];
    A[t] = fmaxf(a + sb[256 + t], 0.f);        __syncthreads();
    a = 0.f; for (int k = 0; k < 256; ++k) a += A[k]  * sw[2*65536 + k*256 + t];
    Bv[t] = R[t] + fmaxf(a + sb[512 + t], 0.f);__syncthreads();   // fx1
    R[t] = Bv[t];                              __syncthreads();
    a = 0.f; for (int k = 0; k < 256; ++k) a += Bv[k] * sw[3*65536 + k*256 + t];
    A[t] = fmaxf(a + sb[768 + t], 0.f);        __syncthreads();
    a = 0.f; for (int k = 0; k < 256; ++k) a += A[k]  * sw[4*65536 + k*256 + t];
    Bv[t] = R[t] + fmaxf(a + sb[1024 + t], 0.f);__syncthreads();  // fx
    red[t] = Bv[t] * Bv[t];                    __syncthreads();
    for (int s = 128; s > 0; s >>= 1) { if (t < s) red[t] += red[t + s]; __syncthreads(); }
    float ssq = red[0];                        __syncthreads();
    a = 0.f; for (int k = 0; k < 256; ++k) a += Bv[k] * proj[k*256 + t];
    expfx[t] = expf(a - 0.5f * ssq);
}

// g[d] = sum_j exp(phi_y[j,d]);  sden = log(g . EXPFX)   (start-dist denominator)
__global__ __launch_bounds__(256) void k_g(
    const float* __restrict__ expl, const float* __restrict__ expfx,
    float* __restrict__ g, float* __restrict__ sden)
{
    const int t = threadIdx.x;
    float a = 0.f;
    for (int j = 0; j < CSTATE; ++j) a += expl[(size_t)j * 256 + t];
    g[t] = a;
    __shared__ float red[256];
    red[t] = a * expfx[t];
    __syncthreads();
    for (int s = 128; s > 0; s >>= 1) { if (t < s) red[t] += red[t + s]; __syncthreads(); }
    if (t == 0) sden[0] = logf(red[0]);
}

// h[n,d] = sum over words in cluster n of exp(phi_term[w,d])  (deterministic gather)
__global__ __launch_bounds__(256) void k_hcl(
    const float* __restrict__ expt, const int* __restrict__ w2s, float* __restrict__ hcl)
{
    const int n = blockIdx.x, t = threadIdx.x;
    float a = 0.f;
    for (int w = 0; w < VWORDS; ++w)
        if ((w2s[w * 32] >> 5) == n) a += expt[(size_t)w * 256 + t];
    hcl[n * 256 + t] = a;
}

// out[row] = log(X[row,:] . vec) - sub ;  vec = vbase (+ cluster offset if clusterVec)
__global__ __launch_bounds__(256) void k_rowlog(
    const float* __restrict__ X, const float* __restrict__ vbase, int clusterVec,
    const float* __restrict__ sub, float* __restrict__ out)
{
    const int row = blockIdx.x, t = threadIdx.x;
    const float* v = vbase + (clusterVec ? (row >> 5) * 256 : 0);
    __shared__ float red[256];
    red[t] = X[(size_t)row * 256 + t] * v[t];
    __syncthreads();
    for (int s = 128; s > 0; s >>= 1) { if (t < s) red[t] += red[t + s]; __syncthreads(); }
    if (t == 0) out[row] = logf(red[0]) - (sub ? sub[0] : 0.f);
}

// emit_bt[b,t,s] = log(exp(phi_fp[c]) . exp(phi_term[w])) - logdenE[c],
// c = cluster(w)*32 + s;  also record cluster id per (b,t).
__global__ __launch_bounds__(256) void k_emit(
    const int* __restrict__ text, const int* __restrict__ w2s,
    const float* __restrict__ expfp, const float* __restrict__ expt,
    const float* __restrict__ logdenE, float* __restrict__ embt, int* __restrict__ clid)
{
    const int bt = blockIdx.x, t = threadIdx.x;
    const int w = text[bt];
    const int n = w2s[w * 32] >> 5;
    if (t == 0) clid[bt] = n;
    const int s = t >> 3, sub = t & 7;
    const float* fr = expfp + (size_t)(n * 32 + s) * 256;
    const float* tr = expt  + (size_t)w * 256;
    float p = 0.f;
    for (int q = 0; q < 32; ++q) { int d = sub + (q << 3); p += fr[d] * tr[d]; }
    __shared__ float red[256];
    red[t] = p;
    __syncthreads();
    if (sub == 0) {
        float ssum = 0.f;
        for (int e = 0; e < 8; ++e) ssum += red[t + e];
        embt[bt * 32 + s] = logf(ssum) - logdenE[n * 32 + s];
    }
}

// Sequential forward scan, one block per batch row. Rank-D contraction form:
//   u = sum_i exp(alpha_i - logden[cp_i] - m) * exp(phi_s[cp_i])
//   alpha'[j] = log(u . exp(phi_y[cc_j])) + m + emit_bt[b,t,j]
// exp(phi_y) block for the step is async-prefetched into LDS while u is built.
__global__ __launch_bounds__(256) void k_scan(
    const int* __restrict__ clid, const float* __restrict__ embt,
    const float* __restrict__ startlp, const float* __restrict__ logden,
    const float* __restrict__ exps, const float* __restrict__ expl,
    float* __restrict__ out)
{
    const int b = blockIdx.x, t = threadIdx.x;
    __shared__ float alpha[32], wts[32], u[256], red[256];
#ifdef USE_ASYNC_LDS
    __shared__ float lcl[32 * 256];   // 32KB: exp(phi_y) rows of current cluster
#endif
    const int n0 = clid[b * TLEN];
    if (t < 32) alpha[t] = startlp[n0 * 32 + t] + embt[(b * TLEN) * 32 + t];
    __syncthreads();

    for (int step = 1; step < TLEN; ++step) {
        const int np = clid[b * TLEN + step - 1];
        const int nc = clid[b * TLEN + step];
        const float* lsrc = expl + (size_t)(nc << 5) * 256;
#ifdef USE_ASYNC_LDS
        // kick off async global->LDS copy of the 32x256 exp(phi_y) block
        #pragma unroll
        for (int q = 0; q < 8; ++q)
            async_copy16(lsrc + q * 1024 + t * 4, lcl + q * 1024 + t * 4);
#endif
        if (t < 32) { float at = alpha[t] - logden[np * 32 + t]; wts[t] = at; red[t] = at; }
        __syncthreads();
        for (int s = 16; s > 0; s >>= 1) { if (t < s) red[t] = fmaxf(red[t], red[t + s]); __syncthreads(); }
        const float m = red[0];
        __syncthreads();
        if (t < 32) wts[t] = expf(wts[t] - m);
        __syncthreads();
        // u[d] = sum_i wts[i] * exp(phi_s)[np*32+i, d]   (coalesced, L2-resident)
        float uu = 0.f;
        const float* eb = exps + (size_t)(np << 5) * 256;
        #pragma unroll 4
        for (int i = 0; i < 32; ++i) uu += wts[i] * eb[(size_t)i * 256 + t];
        u[t] = uu;
        if (step + 1 < TLEN)  // hint next step's exp(phi_s) block toward the caches
            __builtin_prefetch(exps + (size_t)(clid[b * TLEN + step] << 5) * 256 + t * 4, 0, 1);
#ifdef USE_ASYNC_LDS
        WAIT_ASYNC();
#endif
        __syncthreads();
        const int j = t >> 3, sub = t & 7;
#ifdef USE_ASYNC_LDS
        const float* lr = lcl + j * 256;
#else
        const float* lr = lsrc + (size_t)j * 256;
#endif
        float p = 0.f;
        for (int q = 0; q < 32; ++q) { int d = sub + (q << 3); p += u[d] * lr[d]; }
        red[t] = p;
        __syncthreads();
        if (sub == 0) {
            float ssum = 0.f;
            for (int e = 0; e < 8; ++e) ssum += red[t + e];
            alpha[j] = logf(ssum) + m + embt[(b * TLEN + step) * 32 + j];
        }
        __syncthreads();
    }
    // final logsumexp over 32 states
    if (t < 32) red[t] = alpha[t];
    __syncthreads();
    for (int s = 16; s > 0; s >>= 1) { if (t < s) red[t] = fmaxf(red[t], red[t + s]); __syncthreads(); }
    const float m = red[0];
    __syncthreads();
    if (t < 32) red[t] = expf(alpha[t] - m);
    __syncthreads();
    for (int s = 16; s > 0; s >>= 1) { if (t < s) red[t] += red[t + s]; __syncthreads(); }
    if (t == 0) out[b] = m + logf(red[0]);
}

// ---------------------------------------------------------------------------
extern "C" void kernel_launch(void* const* d_in, const int* in_sizes, int n_in,
                              void* d_out, int out_size, void* d_ws, size_t ws_size,
                              hipStream_t stream)
{
    (void)in_sizes; (void)n_in; (void)out_size; (void)ws_size;
    const int*   text      = (const int*)  d_in[0];
    const float* start_emb = (const float*)d_in[1];
    const float* start_w   = (const float*)d_in[2];
    const float* start_b   = (const float*)d_in[3];
    const float* state_emb = (const float*)d_in[4];
    const float* next_emb  = (const float*)d_in[5];
    const float* pre_emb   = (const float*)d_in[6];
    const float* term_w    = (const float*)d_in[7];
    const float* term_b    = (const float*)d_in[8];
    const float* term_emb  = (const float*)d_in[9];
    const float* proj      = (const float*)d_in[10];
    const int*   w2s       = (const int*)  d_in[11];

    float* ws = (float*)d_ws;
    size_t off = 0;
    auto alloc = [&](size_t n) { float* p = ws + off; off += n; return p; };
    float* EXPS   = alloc((size_t)CSTATE * DDIM);   // exp(phi_state)
    float* EXPL   = alloc((size_t)CSTATE * DDIM);   // exp(phi_next)   (lpy)
    float* EXPT   = alloc((size_t)VWORDS * DDIM);   // exp(phi_term)
    float* H1     = alloc((size_t)CSTATE * HDIM);   // MLP hidden (reused)
    float* Y1     = alloc((size_t)CSTATE * HDIM);
    float* FP     = alloc((size_t)CSTATE * HDIM);
    float* EXPFP  = alloc((size_t)CSTATE * DDIM);
    float* G      = alloc(DDIM);
    float* HCL    = alloc((size_t)NCLUS * DDIM);
    float* LOGDEN = alloc(CSTATE);
    float* LOGDENE= alloc(CSTATE);
    float* STARTLP= alloc(CSTATE);
    float* EXPFX  = alloc(DDIM);
    float* SDEN   = alloc(8);
    float* EMBT   = alloc((size_t)BBATCH * TLEN * 32);
    int*   CLID   = (int*)alloc(BBATCH * TLEN);

    const dim3 blk(128);
    const dim3 gC((CSTATE + 31) / 32, 256 / 32);
    const dim3 gV((VWORDS + 31) / 32, 256 / 32);

    // fp = res(res(pre_emb, tw0,tb0, tw1,tb1), tw2,tb2, tw3,tb3)
    wmma_gemm256<<<gC, blk, 0, stream>>>(pre_emb, term_w,            term_b,       nullptr, H1, CSTATE, 0);
    wmma_gemm256<<<gC, blk, 0, stream>>>(H1,      term_w + 65536,    term_b + 256, pre_emb, Y1, CSTATE, 1);
    wmma_gemm256<<<gC, blk, 0, stream>>>(Y1,      term_w + 2*65536,  term_b + 512, nullptr, H1, CSTATE, 0);
    wmma_gemm256<<<gC, blk, 0, stream>>>(H1,      term_w + 3*65536,  term_b + 768, Y1,      FP, CSTATE, 1);
    // exp(log_phi(.)) for all embedding matrices
    wmma_gemm256<<<gC, blk, 0, stream>>>(FP,        proj, nullptr, nullptr, EXPFP, CSTATE, 2);
    wmma_gemm256<<<gC, blk, 0, stream>>>(state_emb, proj, nullptr, nullptr, EXPS,  CSTATE, 2);
    wmma_gemm256<<<gC, blk, 0, stream>>>(next_emb,  proj, nullptr, nullptr, EXPL,  CSTATE, 2);
    wmma_gemm256<<<gV, blk, 0, stream>>>(term_emb,  proj, nullptr, nullptr, EXPT,  VWORDS, 2);

    k_fx<<<1, 256, 0, stream>>>(start_emb, start_w, start_b, proj, EXPFX);
    k_g <<<1, 256, 0, stream>>>(EXPL, EXPFX, G, SDEN);
    k_hcl<<<NCLUS, 256, 0, stream>>>(EXPT, w2s, HCL);

    k_rowlog<<<CSTATE, 256, 0, stream>>>(EXPS,  G,     0, nullptr, LOGDEN);   // trans denominators
    k_rowlog<<<CSTATE, 256, 0, stream>>>(EXPFP, HCL,   1, nullptr, LOGDENE);  // emit denominators
    k_rowlog<<<CSTATE, 256, 0, stream>>>(EXPL,  EXPFX, 0, SDEN,    STARTLP);  // start log-probs

    k_emit<<<BBATCH * TLEN, 256, 0, stream>>>(text, w2s, EXPFP, EXPT, LOGDENE, EMBT, CLID);
    k_scan<<<BBATCH, 256, 0, stream>>>(CLID, EMBT, STARTLP, LOGDEN, EXPS, EXPL, (float*)d_out);
}
